// Mamba2HybridBlock_17033840296583
// MI455X (gfx1250) — matrix operand order repaired and checked by reading, not compiled, not run on hardware
//
#include <hip/hip_runtime.h>

// ---------------- problem constants ----------------
#define B_SZ     4
#define L_SEQ    1024
#define D_MODEL  512
#define D_INNER  1024
#define D_STATE  64
#define NHEADS   16
#define HEADDIM  64
#define CONV_DIM 1152          // D_INNER + 2*D_STATE
#define D_IN_PROJ 2192         // 2*D_INNER + 2*D_STATE + NHEADS
#define NTOK     4096          // B*L
#define INTER_T  2816
#define INTER_M  1536
#define EPS      1e-5f

// ---------------- CDNA5 WMMA types ----------------
typedef __attribute__((ext_vector_type(16))) __bf16 v16bf;
typedef __attribute__((ext_vector_type(8)))  float  v8f;
union ABf { v16bf v; uint4 q[2]; };

// gfx1250 async global->LDS path (ASYNCcnt-tracked), if the toolchain exposes it
#if defined(__has_builtin)
#if __has_builtin(__builtin_amdgcn_global_load_async_to_lds_b128) && \
    __has_builtin(__builtin_amdgcn_s_wait_asynccnt)
#define USE_ASYNC_LDS 1
#endif
#endif
#ifndef USE_ASYNC_LDS
#define USE_ASYNC_LDS 0
#endif

#if USE_ASYNC_LDS
// builtin signature (from compiler diagnostic): param0 = AS1 "int __vector_size__(16)" ptr
typedef int vi4 __attribute__((vector_size(16)));
typedef __attribute__((address_space(1))) vi4 vi4_g;
typedef __attribute__((address_space(3))) vi4 vi4_l;
#endif

__device__ __forceinline__ unsigned short f32_to_bf16(float f) {
  unsigned int u = __float_as_uint(f);
  unsigned int r = u + 0x7FFFu + ((u >> 16) & 1u);   // RNE
  return (unsigned short)(r >> 16);
}

__device__ __forceinline__ float silu_f(float x) {
  return x / (1.0f + expf(-x));
}

__device__ __forceinline__ float block_reduce_sum256(float v, float* sdata) {
  int t = threadIdx.x;
  sdata[t] = v;
  __syncthreads();
  for (int s = 128; s > 0; s >>= 1) {
    if (t < s) sdata[t] += sdata[t + s];
    __syncthreads();
  }
  float r = sdata[0];
  __syncthreads();
  return r;
}

// ---------------- f32 -> bf16 convert ----------------
__global__ void convert_bf16(const float* __restrict__ in,
                             unsigned short* __restrict__ out, size_t n) {
  for (size_t i = (size_t)blockIdx.x * blockDim.x + threadIdx.x; i < n;
       i += (size_t)gridDim.x * blockDim.x)
    out[i] = f32_to_bf16(in[i]);
}

// ---------------- WMMA GEMM: C[M,N] = A[M,K](bf16) @ W[N,K](bf16)^T ----------------
// block tile 128x128x64, 8 wave32 waves (2 in M x 4 in N), each wave 64x32.
// All K used here are multiples of 64.
#define TILE_M 128
#define TILE_N 128
#define TILE_K 64
#define LDSS   72   // row stride (elements); 144B rows keep every b128 access 16B aligned

__global__ __launch_bounds__(256)
void gemm_bf16_wmma(const unsigned short* __restrict__ A,
                    const unsigned short* __restrict__ W,
                    float* __restrict__ C, int M, int N, int K) {
#if USE_ASYNC_LDS
  __shared__ __align__(16) unsigned short As[2][TILE_M * LDSS];
  __shared__ __align__(16) unsigned short Bs[2][TILE_N * LDSS];
#else
  __shared__ __align__(16) unsigned short As[1][TILE_M * LDSS];
  __shared__ __align__(16) unsigned short Bs[1][TILE_N * LDSS];
#endif

  const int tid   = threadIdx.x;
  const int lane  = tid & 31;
  const int wave  = tid >> 5;
  const int waveM = wave & 1;    // 0..1 -> 64 rows
  const int waveN = wave >> 1;   // 0..3 -> 32 cols
  const int row16 = lane & 15;
  const int half  = lane >> 4;

  const int tileM = blockIdx.y * TILE_M;
  const int tileN = blockIdx.x * TILE_N;
  const int nk    = K / TILE_K;

  v8f acc[4][2];
#pragma unroll
  for (int i = 0; i < 4; ++i)
#pragma unroll
    for (int j = 0; j < 2; ++j)
      acc[i][j] = v8f{0.f, 0.f, 0.f, 0.f, 0.f, 0.f, 0.f, 0.f};

#if USE_ASYNC_LDS
  // ---- async double-buffered staging: 8 async-load instructions per stage per wave ----
  auto stage = [&](int buf, int k0) {
#pragma unroll
    for (int t = 0; t < 4; ++t) {
      int idx = tid * 8 + t * 2048;
      int r = idx >> 6, c = idx & 63;
      int gr = tileM + r; if (gr >= M) gr = 0;   // clamp (keeps EXEC/ASYNCcnt uniform)
      int gn = tileN + r; if (gn >= N) gn = 0;
      __builtin_amdgcn_global_load_async_to_lds_b128(
          (vi4_g*)&A[(size_t)gr * K + k0 + c],
          (vi4_l*)&As[buf][r * LDSS + c], 0, 0);
      __builtin_amdgcn_global_load_async_to_lds_b128(
          (vi4_g*)&W[(size_t)gn * K + k0 + c],
          (vi4_l*)&Bs[buf][r * LDSS + c], 0, 0);
    }
  };

  stage(0, 0);
  for (int ks = 0; ks < nk; ++ks) {
    const int buf = ks & 1;
    if (ks + 1 < nk) {
      stage(buf ^ 1, (ks + 1) * TILE_K);
      __builtin_amdgcn_s_wait_asynccnt(8);   // the <=8 newest are the next stage
    } else {
      __builtin_amdgcn_s_wait_asynccnt(0);
    }
    __syncthreads();

    ABf afr[4], bfr[2];
#pragma unroll
    for (int kk = 0; kk < TILE_K; kk += 32) {
#pragma unroll
      for (int i = 0; i < 4; ++i) {
        int ar = waveM * 64 + i * 16 + row16;
        afr[i].q[0] = *(const uint4*)(&As[buf][ar * LDSS + kk + half * 8]);
        afr[i].q[1] = *(const uint4*)(&As[buf][ar * LDSS + kk + 16 + half * 8]);
      }
#pragma unroll
      for (int j = 0; j < 2; ++j) {
        int br = waveN * 32 + j * 16 + row16;
        bfr[j].q[0] = *(const uint4*)(&Bs[buf][br * LDSS + kk + half * 8]);
        bfr[j].q[1] = *(const uint4*)(&Bs[buf][br * LDSS + kk + 16 + half * 8]);
      }
#pragma unroll
      for (int i = 0; i < 4; ++i)
#pragma unroll
        for (int j = 0; j < 2; ++j)
          acc[i][j] = __builtin_amdgcn_wmma_f32_16x16x32_bf16(
              false, afr[i].v, false, bfr[j].v, (short)0, acc[i][j], false, false);
    }
    __syncthreads();   // protect buf from being overwritten two stages later
  }
#else
  // ---- fallback: synchronous staging (global_load -> ds_store) ----
  for (int ks = 0; ks < nk; ++ks) {
    const int k0 = ks * TILE_K;
#pragma unroll
    for (int t = 0; t < 4; ++t) {
      int idx = tid * 8 + t * 2048;
      int r = idx >> 6, c = idx & 63;
      uint4 va = make_uint4(0u, 0u, 0u, 0u);
      uint4 vb = make_uint4(0u, 0u, 0u, 0u);
      int gr = tileM + r;
      int gn = tileN + r;
      if (gr < M) {
        va = *(const uint4*)(&A[(size_t)gr * K + k0 + c]);
        if (k0 + TILE_K < K)
          __builtin_prefetch(&A[(size_t)gr * K + k0 + TILE_K + c], 0, 1);
      }
      if (gn < N) {
        vb = *(const uint4*)(&W[(size_t)gn * K + k0 + c]);
        if (k0 + TILE_K < K)
          __builtin_prefetch(&W[(size_t)gn * K + k0 + TILE_K + c], 0, 1);
      }
      *(uint4*)(&As[0][r * LDSS + c]) = va;
      *(uint4*)(&Bs[0][r * LDSS + c]) = vb;
    }
    __syncthreads();

    ABf afr[4], bfr[2];
#pragma unroll
    for (int kk = 0; kk < TILE_K; kk += 32) {
#pragma unroll
      for (int i = 0; i < 4; ++i) {
        int ar = waveM * 64 + i * 16 + row16;
        afr[i].q[0] = *(const uint4*)(&As[0][ar * LDSS + kk + half * 8]);
        afr[i].q[1] = *(const uint4*)(&As[0][ar * LDSS + kk + 16 + half * 8]);
      }
#pragma unroll
      for (int j = 0; j < 2; ++j) {
        int br = waveN * 32 + j * 16 + row16;
        bfr[j].q[0] = *(const uint4*)(&Bs[0][br * LDSS + kk + half * 8]);
        bfr[j].q[1] = *(const uint4*)(&Bs[0][br * LDSS + kk + 16 + half * 8]);
      }
#pragma unroll
      for (int i = 0; i < 4; ++i)
#pragma unroll
        for (int j = 0; j < 2; ++j)
          acc[i][j] = __builtin_amdgcn_wmma_f32_16x16x32_bf16(
              false, afr[i].v, false, bfr[j].v, (short)0, acc[i][j], false, false);
    }
    __syncthreads();
  }
#endif

  // ---- epilogue: 16x16 f32 C/D layout: VGPR r -> M = half*8 + r, N = lane&15 ----
#pragma unroll
  for (int i = 0; i < 4; ++i)
#pragma unroll
    for (int j = 0; j < 2; ++j) {
      int col = tileN + waveN * 32 + j * 16 + row16;
      if (col < N) {
#pragma unroll
        for (int r = 0; r < 8; ++r) {
          int row = tileM + waveM * 64 + i * 16 + half * 8 + r;
          if (row < M) C[(size_t)row * N + col] = acc[i][j][r];
        }
      }
    }
}

// ---------------- dt = softplus(zx[..., -NH:] + dt_bias) ----------------
__global__ void compute_dt(const float* __restrict__ zx,
                           const float* __restrict__ dt_bias,
                           float* __restrict__ dt) {
  int i = blockIdx.x * blockDim.x + threadIdx.x;
  if (i < NTOK * NHEADS) {
    int row = i >> 4, h = i & 15;
    float x = zx[(size_t)row * D_IN_PROJ + (D_IN_PROJ - NHEADS) + h] + dt_bias[h];
    dt[i] = (x > 20.f) ? x : log1pf(expf(x));
  }
}

// ---------------- causal depthwise conv (width 4) + silu ----------------
__global__ void conv_silu(const float* __restrict__ zx,
                          const float* __restrict__ conv_w,
                          const float* __restrict__ conv_b,
                          float* __restrict__ xBC) {
  size_t total = (size_t)NTOK * CONV_DIM;
  for (size_t i = (size_t)blockIdx.x * blockDim.x + threadIdx.x; i < total;
       i += (size_t)gridDim.x * blockDim.x) {
    int c = (int)(i % CONV_DIM);
    size_t row = i / CONV_DIM;
    int b = (int)(row / L_SEQ), l = (int)(row % L_SEQ);
    float acc = conv_b[c];
#pragma unroll
    for (int tap = 0; tap < 4; ++tap) {
      int l2 = l - 3 + tap;
      if (l2 >= 0)
        acc += zx[((size_t)b * L_SEQ + l2) * D_IN_PROJ + D_INNER + c] * conv_w[c * 4 + tap];
    }
    xBC[i] = silu_f(acc);
  }
}

// ---------------- SSM sequential scan: one block per (b,h) ----------------
// state s[HEADDIM=64][D_STATE=64] striped: thread t owns p = t>>2, n = (t&3)*16 .. +16
__global__ __launch_bounds__(256)
void ssm_scan(const float* __restrict__ xBC, const float* __restrict__ dt,
              const float* __restrict__ A_log, const float* __restrict__ Dp,
              float* __restrict__ y) {
  int b = blockIdx.x >> 4;
  int h = blockIdx.x & 15;
  __shared__ float xs[64], Bsh[64], Csh[64];
  __shared__ float dts;

  const float A  = -expf(A_log[h]);
  const float Dh = Dp[h];
  const int t  = threadIdx.x;
  const int p  = t >> 2;
  const int nb = (t & 3) << 4;

  float s[16];
#pragma unroll
  for (int jj = 0; jj < 16; ++jj) s[jj] = 0.f;

  for (int l = 0; l < L_SEQ; ++l) {
    size_t row = (size_t)b * L_SEQ + l;
    if (t < 64)        xs[t]        = xBC[row * CONV_DIM + h * HEADDIM + t];
    else if (t < 128)  Bsh[t - 64]  = xBC[row * CONV_DIM + D_INNER + (t - 64)];
    else if (t < 192)  Csh[t - 128] = xBC[row * CONV_DIM + D_INNER + D_STATE + (t - 128)];
    else if (t == 192) dts          = dt[row * NHEADS + h];
    __syncthreads();

    float dtv = dts;
    float dA  = expf(dtv * A);
    float xd  = dtv * xs[p];
    float yp  = 0.f;
#pragma unroll
    for (int jj = 0; jj < 16; ++jj) {
      s[jj] = s[jj] * dA + xd * Bsh[nb + jj];
      yp += s[jj] * Csh[nb + jj];
    }
    yp += __shfl_xor(yp, 1, 32);
    yp += __shfl_xor(yp, 2, 32);
    if ((t & 3) == 0)
      y[row * D_INNER + h * HEADDIM + p] = yp + Dh * xs[p];
    __syncthreads();
  }
}

// ---------------- g = rmsnorm(y * silu(z)) * norm_w  -> bf16 ----------------
__global__ __launch_bounds__(256)
void gated_rmsnorm(const float* __restrict__ y, const float* __restrict__ zx,
                   const float* __restrict__ norm_w,
                   unsigned short* __restrict__ out) {
  __shared__ float red[256];
  int row = blockIdx.x, t = threadIdx.x;
  float g[4];
  float ss = 0.f;
#pragma unroll
  for (int k = 0; k < 4; ++k) {
    int c = t + (k << 8);
    float z  = zx[(size_t)row * D_IN_PROJ + c];
    float yy = y[(size_t)row * D_INNER + c];
    float gv = yy * silu_f(z);
    g[k] = gv;
    ss += gv * gv;
  }
  float tot = block_reduce_sum256(ss, red);
  float scale = rsqrtf(tot / (float)D_INNER + EPS);
#pragma unroll
  for (int k = 0; k < 4; ++k) {
    int c = t + (k << 8);
    out[(size_t)row * D_INNER + c] = f32_to_bf16(g[k] * scale * norm_w[c]);
  }
}

// ---------------- out = rmsnorm(a + b), row length len (512 or 1024) ----------------
__global__ __launch_bounds__(256)
void add_rmsnorm(const float* __restrict__ a, const float* __restrict__ b,
                 float* __restrict__ out, int len) {
  __shared__ float red[256];
  int row = blockIdx.x, t = threadIdx.x;
  int per = len >> 8;  // <= 4
  float v[4];
  float ss = 0.f;
  for (int k = 0; k < per; ++k) {
    int c = t + (k << 8);
    float x = a[(size_t)row * len + c] + b[(size_t)row * len + c];
    v[k] = x;
    ss += x * x;
  }
  float tot = block_reduce_sum256(ss, red);
  float scale = rsqrtf(tot / (float)len + EPS);
  for (int k = 0; k < per; ++k)
    out[(size_t)row * len + t + (k << 8)] = v[k] * scale;
}

// ---------------- transpose [Bt,R,C] -> [Bt,C,R]; f32 + bf16 outputs ----------------
__global__ void transpose_f32_bf16(const float* __restrict__ in,
                                   float* __restrict__ out_f,
                                   unsigned short* __restrict__ out_b,
                                   int R, int C) {
  __shared__ float tile[32][33];
  int bt = blockIdx.z;
  int c0 = blockIdx.x * 32, r0 = blockIdx.y * 32;
  const float* inp = in + (size_t)bt * R * C;
  for (int i = threadIdx.y; i < 32; i += 8) {
    int r = r0 + i, c = c0 + threadIdx.x;
    tile[i][threadIdx.x] = (r < R && c < C) ? inp[(size_t)r * C + c] : 0.f;
  }
  __syncthreads();
  float* of = out_f + (size_t)bt * R * C;
  unsigned short* ob = out_b + (size_t)bt * R * C;
  for (int i = threadIdx.y; i < 32; i += 8) {
    int c = c0 + i, r = r0 + threadIdx.x;
    if (c < C && r < R) {
      float v = tile[threadIdx.x][i];
      of[(size_t)c * R + r] = v;
      ob[(size_t)c * R + r] = f32_to_bf16(v);
    }
  }
}

// ---------------- mid = silu(G[:, :inter]) * G[:, inter:] -> bf16 ----------------
__global__ void swiglu_combine(const float* __restrict__ G,
                               unsigned short* __restrict__ out,
                               int rows, int inter) {
  size_t total = (size_t)rows * inter;
  for (size_t i = (size_t)blockIdx.x * blockDim.x + threadIdx.x; i < total;
       i += (size_t)gridDim.x * blockDim.x) {
    size_t r = i / inter, c = i % inter;
    float g = G[r * (size_t)(2 * inter) + c];
    float u = G[r * (size_t)(2 * inter) + inter + c];
    out[i] = f32_to_bf16(silu_f(g) * u);
  }
}

// ---------------- host orchestration ----------------
extern "C" void kernel_launch(void* const* d_in, const int* in_sizes, int n_in,
                              void* d_out, int out_size, void* d_ws, size_t ws_size,
                              hipStream_t stream) {
  (void)in_sizes; (void)n_in; (void)out_size; (void)ws_size;

  const float* hidden   = (const float*)d_in[0];
  const float* in_proj  = (const float*)d_in[1];
  const float* conv_w   = (const float*)d_in[2];
  const float* conv_b   = (const float*)d_in[3];
  const float* dt_bias  = (const float*)d_in[4];
  const float* A_log    = (const float*)d_in[5];
  const float* Dp       = (const float*)d_in[6];
  const float* norm_w   = (const float*)d_in[7];
  const float* out_proj = (const float*)d_in[8];
  const float* gu_t     = (const float*)d_in[9];
  const float* down_t   = (const float*)d_in[10];
  const float* gu_m     = (const float*)d_in[11];
  const float* down_m   = (const float*)d_in[12];
  float* out = (float*)d_out;

  char* base = (char*)d_ws;
  size_t off = 0;
  auto carve = [&](size_t bytes) -> char* {
    char* r = base + off;
    off += (bytes + 255) & ~(size_t)255;
    return r;
  };

  unsigned short* u_bf   = (unsigned short*)carve((size_t)NTOK * D_MODEL * 2);
  unsigned short* wi_bf  = (unsigned short*)carve((size_t)D_IN_PROJ * D_MODEL * 2);
  unsigned short* wo_bf  = (unsigned short*)carve((size_t)D_MODEL * D_INNER * 2);
  unsigned short* wgt_bf = (unsigned short*)carve((size_t)2 * INTER_T * L_SEQ * 2);
  unsigned short* wdt_bf = (unsigned short*)carve((size_t)L_SEQ * INTER_T * 2);
  unsigned short* wgm_bf = (unsigned short*)carve((size_t)2 * INTER_M * D_MODEL * 2);
  unsigned short* wdm_bf = (unsigned short*)carve((size_t)D_MODEL * INTER_M * 2);
  float* zx   = (float*)carve((size_t)NTOK * D_IN_PROJ * 4);
  float* dtb  = (float*)carve((size_t)NTOK * NHEADS * 4);
  float* xBC  = (float*)carve((size_t)NTOK * CONV_DIM * 4);
  float* ybuf = (float*)carve((size_t)NTOK * D_INNER * 4);
  unsigned short* g_bf = (unsigned short*)carve((size_t)NTOK * D_INNER * 2);
  float* mo = (float*)carve((size_t)NTOK * D_MODEL * 4);
  float* h1 = (float*)carve((size_t)NTOK * D_MODEL * 4);
  float* xt = (float*)carve((size_t)B_SZ * D_MODEL * L_SEQ * 4);
  unsigned short* xt_bf = (unsigned short*)carve((size_t)B_SZ * D_MODEL * L_SEQ * 2);
  float* G1 = (float*)carve((size_t)B_SZ * D_MODEL * 2 * INTER_T * 4);
  unsigned short* mid_bf = (unsigned short*)carve((size_t)B_SZ * D_MODEL * INTER_T * 2);
  float* S1 = (float*)carve((size_t)B_SZ * D_MODEL * L_SEQ * 4);
  float* x2 = (float*)carve((size_t)B_SZ * D_MODEL * L_SEQ * 4);
  float* h2 = (float*)carve((size_t)NTOK * D_MODEL * 4);
  unsigned short* h2_bf = (unsigned short*)carve((size_t)NTOK * D_MODEL * 2);
  float* G2 = (float*)carve((size_t)NTOK * 2 * INTER_M * 4);
  unsigned short* midm_bf = (unsigned short*)carve((size_t)NTOK * INTER_M * 2);
  float* S2 = (float*)carve((size_t)NTOK * D_MODEL * 4);

  auto cvt = [&](const float* src, unsigned short* dst, size_t n) {
    int blocks = (int)((n + 255) / 256);
    if (blocks > 8192) blocks = 8192;
    convert_bf16<<<blocks, 256, 0, stream>>>(src, dst, n);
  };
  auto gemm = [&](const unsigned short* A, const unsigned short* W, float* Cc,
                  int M, int N, int K) {
    dim3 g((N + TILE_N - 1) / TILE_N, (M + TILE_M - 1) / TILE_M);
    gemm_bf16_wmma<<<g, 256, 0, stream>>>(A, W, Cc, M, N, K);
  };

  // weight / activation precision conversion
  cvt(hidden,   u_bf,   (size_t)NTOK * D_MODEL);
  cvt(in_proj,  wi_bf,  (size_t)D_IN_PROJ * D_MODEL);
  cvt(out_proj, wo_bf,  (size_t)D_MODEL * D_INNER);
  cvt(gu_t,     wgt_bf, (size_t)2 * INTER_T * L_SEQ);
  cvt(down_t,   wdt_bf, (size_t)L_SEQ * INTER_T);
  cvt(gu_m,     wgm_bf, (size_t)2 * INTER_M * D_MODEL);
  cvt(down_m,   wdm_bf, (size_t)D_MODEL * INTER_M);

  // ---- mamba2 ----
  gemm(u_bf, wi_bf, zx, NTOK, D_IN_PROJ, D_MODEL);
  compute_dt<<<(NTOK * NHEADS + 255) / 256, 256, 0, stream>>>(zx, dt_bias, dtb);
  conv_silu<<<4096, 256, 0, stream>>>(zx, conv_w, conv_b, xBC);
  ssm_scan<<<B_SZ * NHEADS, 256, 0, stream>>>(xBC, dtb, A_log, Dp, ybuf);
  gated_rmsnorm<<<NTOK, 256, 0, stream>>>(ybuf, zx, norm_w, g_bf);
  gemm(g_bf, wo_bf, mo, NTOK, D_MODEL, D_INNER);
  add_rmsnorm<<<NTOK, 256, 0, stream>>>(hidden, mo, h1, D_MODEL);

  // ---- time-mixing swiglu (operates on transposed [B, D, L]) ----
  transpose_f32_bf16<<<dim3(D_MODEL / 32, L_SEQ / 32, B_SZ), dim3(32, 8), 0, stream>>>(
      h1, xt, xt_bf, L_SEQ, D_MODEL);
  gemm(xt_bf, wgt_bf, G1, B_SZ * D_MODEL, 2 * INTER_T, L_SEQ);
  swiglu_combine<<<4096, 256, 0, stream>>>(G1, mid_bf, B_SZ * D_MODEL, INTER_T);
  gemm(mid_bf, wdt_bf, S1, B_SZ * D_MODEL, L_SEQ, INTER_T);
  add_rmsnorm<<<B_SZ * D_MODEL, 256, 0, stream>>>(xt, S1, x2, L_SEQ);

  // ---- channel-mixing swiglu (back to [B, L, D]) ----
  transpose_f32_bf16<<<dim3(L_SEQ / 32, D_MODEL / 32, B_SZ), dim3(32, 8), 0, stream>>>(
      x2, h2, h2_bf, D_MODEL, L_SEQ);
  gemm(h2_bf, wgm_bf, G2, NTOK, 2 * INTER_M, D_MODEL);
  swiglu_combine<<<4096, 256, 0, stream>>>(G2, midm_bf, NTOK, INTER_M);
  gemm(midm_bf, wdm_bf, S2, NTOK, D_MODEL, INTER_M);
  add_rmsnorm<<<NTOK, 256, 0, stream>>>(h2, S2, out, D_MODEL);
}
